// Net_56221121905185
// MI455X (gfx1250) — matrix operand order repaired and checked by reading, not compiled, not run on hardware
//
#include <hip/hip_runtime.h>
#include <hip/hip_bf16.h>

#define NN   10000
#define EE   160000
#define GG   64
#define TT   5
#define FIN  75
#define ED   50
#define FOUT 15
#define KPAD 96    // FIN padded to 3x32 for bf16 WMMA K-steps
#define CPAD 80    // per-tower feature columns padded to 5x16 N-tiles
#define NT5  (CPAD / 16)      // 5 N-tiles per tower
#define NTILES (TT * NT5)     // 25 (tower, N-tile) pairs
#define ETILES (EE / 16)      // 10000 edge tiles

typedef __bf16 bf16_t;
typedef __attribute__((ext_vector_type(16))) bf16_t v16bf;
typedef __attribute__((ext_vector_type(8)))  float  v8f;

// ---------------- small helpers ----------------

// Branchless monotonic float <-> ordered-unsigned mapping:
// float compare == unsigned compare, so segment min/max become atomicMin/Max u32.
__device__ __forceinline__ unsigned f2key(float f) {
  unsigned b = __float_as_uint(f);
  unsigned mask = (unsigned)((int)b >> 31) | 0x80000000u;
  return b ^ mask;
}
__device__ __forceinline__ float key2f(unsigned k) {
  unsigned s = (unsigned)((int)k >> 31);          // all-ones if key's top bit set
  unsigned mask = (~s) | 0x80000000u;
  return __uint_as_float(k ^ mask);
}

__device__ __forceinline__ float softplusf(float x) {
  return (x > 20.f) ? x : log1pf(expf(x));
}
__device__ __forceinline__ float uaf_act(float v, float p0, float p1, float p2, float p3, float p4) {
  return softplusf(p0 * (v + p1) + p2 * v * v) - softplusf(p3 * (v - p1)) + p4;
}

// Map feature index k in [0,96) to its position inside the A-fragment-swizzled
// node row, such that the 16 bf16 values a lane needs for WMMA K-tile kk and
// half-wave h are contiguous at [kk*32 + h*16, +16).
// ISA A 16x32 layout: pair p -> K = 2p + 8h (p<4) | 16 + 2(p-4) + 8h (p>=4).
__device__ __forceinline__ int hswz(int k) {
  int kk = k >> 5, kr = k & 31;
  int hi = (kr >= 16) ? 1 : 0;      // p>=4 group
  int q  = kr - 16 * hi;            // 0..15
  int h  = q >> 3;
  int j  = q & 1;
  int p  = ((q & 7) >> 1) + 4 * hi;
  return kk * 32 + h * 16 + 2 * p + j;
}

__global__ void k_fill(float* p, long n, float v) {
  long i = (long)blockIdx.x * blockDim.x + threadIdx.x;
  if (i < n) p[i] = v;
}
__global__ void k_fillu(unsigned* p, long n, unsigned v) {
  long i = (long)blockIdx.x * blockDim.x + threadIdx.x;
  if (i < n) p[i] = v;
}

// ---------------- stage 0: gather h0 (swizzled bf16 copy), degree, avg_log ----

__global__ void k_gather_h0(const float* __restrict__ node_emb, const int* __restrict__ x,
                            float* __restrict__ h, bf16_t* __restrict__ hb) {
  long i = (long)blockIdx.x * blockDim.x + threadIdx.x;
  if (i >= (long)NN * KPAD) return;
  int n = (int)(i / KPAD), k = (int)(i % KPAD);
  float v = 0.f;
  if (k < FIN) {
    v = node_emb[(long)x[n] * FIN + k];
    h[(long)n * FIN + k] = v;
  }
  hb[(long)n * KPAD + hswz(k)] = (bf16_t)v;
}

__global__ void k_deg(const int* __restrict__ dst, float* __restrict__ deg) {
  long i = (long)blockIdx.x * blockDim.x + threadIdx.x;
  if (i < EE) atomicAdd(&deg[dst[i]], 1.f);
}

__global__ void k_avglog(const float* __restrict__ deg, float* __restrict__ acc) {
  long i = (long)blockIdx.x * blockDim.x + threadIdx.x;
  if (i < NN) atomicAdd(acc, logf(deg[i] + 1.f));
}
__global__ void k_scale1(float* p, float s) { p[0] *= s; }

// ---------------- per-layer prep: fragment-swizzled bf16 weights + c3 table ----

// Wbs layout: [which(2)][t(5)][n5(5)][kk(3)][lane(32)][16 bf16]  (contiguous per lane)
// ISA B 32x16 layout: lane -> col N = lane%16, half = lane/16; pair p -> K = 2p + 8*half (p<4) | 16 + 2(p-4) + 8*half (p>=4).
__global__ void k_prep_w(const float* __restrict__ Wpre_l, bf16_t* __restrict__ Wbs) {
  long i = (long)blockIdx.x * blockDim.x + threadIdx.x;
  const long total = 2L * TT * NT5 * 3 * 32 * 16;
  if (i >= total) return;
  int e     = (int)(i & 15);
  int lane  = (int)((i >> 4) & 31);
  int kk    = (int)((i >> 9) % 3);
  int n5    = (int)((i / (512 * 3)) % NT5);
  int t     = (int)((i / (512 * 3 * NT5)) % TT);
  int which = (int)(i / (512L * 3 * NT5 * TT));
  int half = lane >> 4;
  int col  = n5 * 16 + (lane & 15);
  int p = e >> 1, j = e & 1;
  int k = 32 * kk + ((p < 4) ? (2 * p + 8 * half + j) : (16 + 2 * (p - 4) + 8 * half + j));
  float v = 0.f;
  if (k < FIN && col < FIN)
    v = Wpre_l[(long)t * (3 * FIN) * FIN + (long)(which * FIN + k) * FIN + col];
  Wbs[i] = (bf16_t)v;
}

// c3[a][t][CPAD] = (edge_emb[a] @ We + be) @ W3[t] + bpre[t]   (4 distinct edge types)
__global__ void k_c3(const float* __restrict__ edge_emb, const float* __restrict__ We_l,
                     const float* __restrict__ be_l, const float* __restrict__ Wpre_l,
                     const float* __restrict__ bpre_l, float* __restrict__ c3) {
  __shared__ float eemb[4][FIN];
  for (int idx = threadIdx.x; idx < 4 * FIN; idx += blockDim.x) {
    int a = idx / FIN, f = idx % FIN;
    float acc = be_l[f];
    for (int j = 0; j < ED; ++j) acc += edge_emb[a * ED + j] * We_l[(long)j * FIN + f];
    eemb[a][f] = acc;
  }
  __syncthreads();
  for (int idx = threadIdx.x; idx < 4 * TT * CPAD; idx += blockDim.x) {
    int a = idx / (TT * CPAD), t = (idx / CPAD) % TT, f = idx % CPAD;
    float acc = 0.f;
    if (f < FIN) {
      acc = bpre_l[t * FIN + f];
      for (int c = 0; c < FIN; ++c)
        acc += eemb[a][c] * Wpre_l[(long)t * (3 * FIN) * FIN + (long)(2 * FIN + c) * FIN + f];
    }
    c3[idx] = acc;
  }
}

// ---------------- fused edge-message WMMA + segment aggregation ----------------
// Each wave owns one (tower, N-tile) pair: its 6 weight fragments stay in VGPRs.
// It streams over 16-edge tiles: 6x 32B A-fragment loads (swizzled hb rows),
// 6x v_wmma_f32_16x16x32_bf16 into one f32 accumulator, then +c3[attr] (L2-hot
// table load) and a fully branchless atomic sum/sum^2/min/max scatter into
// per-dst aggregates (the message matrix never touches HBM).
__global__ __launch_bounds__(256) void k_edge_msgs(
    const bf16_t* __restrict__ hb, const bf16_t* __restrict__ Wbs,
    const float* __restrict__ c3,
    const int* __restrict__ src, const int* __restrict__ dst,
    const int* __restrict__ attr,
    float* __restrict__ Ssum, float* __restrict__ Ssq,
    unsigned* __restrict__ Smin, unsigned* __restrict__ Smax) {
  const int lane = threadIdx.x & 31;
  const int wave = threadIdx.x >> 5;
  const int nt = blockIdx.y;              // 0..24
  const int t = nt / NT5, n5 = nt % NT5;
  const int half = lane >> 4;
  const int m16 = lane & 15;
  const int col = n5 * 16 + m16;

  // register-resident weight fragments: [which][t][n5][kk][lane][16]
  v16bf B1[3], B2[3];
#pragma unroll
  for (int kk = 0; kk < 3; ++kk) {
    B1[kk] = *(const v16bf*)(Wbs + ((((0L * TT + t) * NT5 + n5) * 3 + kk) * 32 + lane) * 16);
    B2[kk] = *(const v16bf*)(Wbs + ((((1L * TT + t) * NT5 + n5) * 3 + kk) * 32 + lane) * 16);
  }
  // per-lane base into the c3 table for this wave's fixed (t, col)
  const float* c3tc = c3 + t * CPAD + col;

  for (int tile = blockIdx.x * 8 + wave; tile < ETILES; tile += gridDim.x * 8) {
    const int e0 = tile * 16;
    const int nd = dst[e0 + m16];
    const int ns = src[e0 + m16];
    const bf16_t* rowD = hb + (size_t)nd * KPAD;
    const bf16_t* rowS = hb + (size_t)ns * KPAD;
    v8f acc = {};
#pragma unroll
    for (int kk = 0; kk < 3; ++kk) {
      v16bf a = *(const v16bf*)(rowD + kk * 32 + half * 16);
      acc = __builtin_amdgcn_wmma_f32_16x16x32_bf16(false, a, false, B1[kk],
                                                    (short)0, acc, false, false);
    }
#pragma unroll
    for (int kk = 0; kk < 3; ++kk) {
      v16bf a = *(const v16bf*)(rowS + kk * 32 + half * 16);
      acc = __builtin_amdgcn_wmma_f32_16x16x32_bf16(false, a, false, B2[kk],
                                                    (short)0, acc, false, false);
    }
    if (col < FIN) {
#pragma unroll
      for (int r = 0; r < 8; ++r) {
        const int e = e0 + r + 8 * half;     // D layout: VGPR r -> edge row r + 8*half
        const int dn = dst[e];
        const int at = attr[e];
        const float v = acc[r] + c3tc[at * (TT * CPAD)];
        const unsigned kv = f2key(v);
        const size_t o = ((size_t)dn * TT + t) * CPAD + col;
        atomicAdd(Ssum + o, v);
        atomicAdd(Ssq + o, v * v);
        atomicMax(Smax + o, kv);
        atomicMin(Smin + o, kv);
      }
    }
  }
}

// ---------------- per-node: build PNA features, Wpost + Wlin GEMVs, BN partial sums ----
__global__ __launch_bounds__(128) void k_node_post(
    const float* __restrict__ h, const float* __restrict__ Ssum, const float* __restrict__ Ssq,
    const unsigned* __restrict__ Smin, const unsigned* __restrict__ Smax,
    const float* __restrict__ deg, const float* __restrict__ avglog,
    const float* __restrict__ Wpost_l, const float* __restrict__ bpost_l,
    const float* __restrict__ Wlin_l, const float* __restrict__ blin_l,
    float* __restrict__ outpre, float* __restrict__ bnsum, float* __restrict__ bnsumsq) {
  __shared__ float feat[TT][13 * FIN];   // [h(75) | agg(300) | agg*amp(300) | agg/amp(300)]
  __shared__ float tmp[TT * FOUT];
  const int n = blockIdx.x;
  const float dg = deg[n];
  const float degc = fmaxf(dg, 1.f);
  const float amp = logf(degc + 1.f) / avglog[0];
  const bool empty = (dg == 0.f);

  for (int idx = threadIdx.x; idx < TT * FIN; idx += blockDim.x)
    feat[idx / FIN][idx % FIN] = h[(long)n * FIN + idx % FIN];
  for (int idx = threadIdx.x; idx < TT * 4 * FIN; idx += blockDim.x) {
    int t = idx / (4 * FIN), g = idx % (4 * FIN);
    int grp = g / FIN, f = g % FIN;
    size_t o = ((size_t)n * TT + t) * CPAD + f;
    float s = Ssum[o], ss = Ssq[o];
    float mnv = key2f(Smin[o]), mxv = key2f(Smax[o]);
    float mean = s / degc;
    float stdv = sqrtf(fmaxf(ss / degc - mean * mean, 0.f) + 1e-5f);
    if (empty) { mnv = 0.f; mxv = 0.f; }
    float a = (grp == 0) ? mean : (grp == 1) ? mnv : (grp == 2) ? mxv : stdv;
    feat[t][FIN + g] = a;
    feat[t][5 * FIN + g] = a * amp;
    feat[t][9 * FIN + g] = a / amp;
  }
  __syncthreads();
  for (int j = threadIdx.x; j < TT * FOUT; j += blockDim.x) {
    int t = j / FOUT, o = j % FOUT;
    float acc = bpost_l[t * FOUT + o];
    const float* Wp = Wpost_l + (size_t)t * (13 * FIN) * FOUT + o;
    const float* fv = feat[t];
    for (int c = 0; c < 13 * FIN; ++c) acc += fv[c] * Wp[(size_t)c * FOUT];
    tmp[j] = acc;
  }
  __syncthreads();
  for (int j = threadIdx.x; j < FIN; j += blockDim.x) {
    float acc = blin_l[j];
    for (int c = 0; c < FIN; ++c) acc += tmp[c] * Wlin_l[(size_t)c * FIN + j];
    outpre[(long)n * FIN + j] = acc;
    atomicAdd(&bnsum[j], acc);
    atomicAdd(&bnsumsq[j], acc * acc);
  }
}

__global__ void k_bnstats(const float* bnsum, const float* bnsumsq, float* mu, float* rstd) {
  int j = threadIdx.x;
  if (j < FIN) {
    float m = bnsum[j] / (float)NN;
    float v = bnsumsq[j] / (float)NN - m * m;
    mu[j] = m;
    rstd[j] = rsqrtf(v + 1e-5f);
  }
}

__global__ void k_bn_act(const float* __restrict__ outpre, const float* __restrict__ mu,
                         const float* __restrict__ rstd, const float* __restrict__ gam,
                         const float* __restrict__ bet, const float* __restrict__ uafp,
                         float* __restrict__ h, bf16_t* __restrict__ hb) {
  long i = (long)blockIdx.x * blockDim.x + threadIdx.x;
  if (i >= (long)NN * FIN) return;
  int n = (int)(i / FIN), f = (int)(i % FIN);
  float v = (outpre[i] - mu[f]) * rstd[f] * gam[f] + bet[f];
  float a = uaf_act(v, uafp[0], uafp[1], uafp[2], uafp[3], uafp[4]);
  h[i] = a;
  hb[(long)n * KPAD + hswz(f)] = (bf16_t)a;   // pad slots (k>=75) stay zero from k_gather_h0
}

// ---------------- pooling + MLP head ----------------

__global__ void k_pool(const float* __restrict__ h, const int* __restrict__ batch,
                       float* __restrict__ pooled) {
  long i = (long)blockIdx.x * blockDim.x + threadIdx.x;
  if (i >= (long)NN * FIN) return;
  int n = (int)(i / FIN), f = (int)(i % FIN);
  atomicAdd(&pooled[(long)batch[n] * FIN + f], h[i]);
}

__global__ __launch_bounds__(64) void k_mlp(const float* __restrict__ pooled,
                                            const float* __restrict__ mW1, const float* __restrict__ mb1,
                                            const float* __restrict__ mW2, const float* __restrict__ mb2,
                                            const float* __restrict__ mW3, const float* __restrict__ mb3,
                                            const float* __restrict__ uafp, float* __restrict__ out) {
  __shared__ float z1[50];
  __shared__ float z2[25];
  const int g = blockIdx.x, j = threadIdx.x;
  const float p0 = uafp[0], p1 = uafp[1], p2 = uafp[2], p3 = uafp[3], p4 = uafp[4];
  if (j < 50) {
    float acc = mb1[j];
    for (int c = 0; c < FIN; ++c) acc += pooled[(long)g * FIN + c] * mW1[c * 50 + j];
    z1[j] = uaf_act(acc, p0, p1, p2, p3, p4);
  }
  __syncthreads();
  if (j < 25) {
    float acc = mb2[j];
    for (int c = 0; c < 50; ++c) acc += z1[c] * mW2[c * 25 + j];
    z2[j] = uaf_act(acc, p0, p1, p2, p3, p4);
  }
  __syncthreads();
  if (j == 0) {
    float acc = mb3[0];
    for (int c = 0; c < 25; ++c) acc += z2[c] * mW3[c];
    out[g] = acc;
  }
}

// ---------------- host-side launcher ----------------

static inline unsigned gdiv(long n, int b) { return (unsigned)((n + b - 1) / b); }

extern "C" void kernel_launch(void* const* d_in, const int* in_sizes, int n_in,
                              void* d_out, int out_size, void* d_ws, size_t ws_size,
                              hipStream_t stream) {
  (void)in_sizes; (void)n_in; (void)out_size; (void)ws_size;
  const int* x      = (const int*)d_in[0];
  const int* eidx   = (const int*)d_in[1];
  const int* eattr  = (const int*)d_in[2];
  const int* batch  = (const int*)d_in[3];
  const float* node_emb = (const float*)d_in[4];
  const float* edge_emb = (const float*)d_in[5];
  const float* We   = (const float*)d_in[6];
  const float* be   = (const float*)d_in[7];
  const float* Wpre = (const float*)d_in[8];
  const float* bpre = (const float*)d_in[9];
  const float* Wpost = (const float*)d_in[10];
  const float* bpost = (const float*)d_in[11];
  const float* Wlin = (const float*)d_in[12];
  const float* blin = (const float*)d_in[13];
  const float* gam  = (const float*)d_in[14];
  const float* bet  = (const float*)d_in[15];
  const float* uafp = (const float*)d_in[16];
  const float* mW1 = (const float*)d_in[17];
  const float* mb1 = (const float*)d_in[18];
  const float* mW2 = (const float*)d_in[19];
  const float* mb2 = (const float*)d_in[20];
  const float* mW3 = (const float*)d_in[21];
  const float* mb3 = (const float*)d_in[22];
  float* out = (float*)d_out;

  const int* srcp = eidx;
  const int* dstp = eidx + EE;

  // bump-allocate workspace
  char* w = (char*)d_ws;
  size_t off = 0;
  auto take = [&](size_t bytes) -> void* {
    void* p = w + off;
    off = (off + bytes + 255) & ~(size_t)255;
    return p;
  };
  float*    h      = (float*)   take((size_t)NN * FIN * 4);
  bf16_t*   hb     = (bf16_t*)  take((size_t)NN * KPAD * 2);
  bf16_t*   Wbs    = (bf16_t*)  take((size_t)2 * TT * NT5 * 3 * 32 * 16 * 2);
  float*    c3     = (float*)   take((size_t)4 * TT * CPAD * 4);
  float*    Ssum   = (float*)   take((size_t)NN * TT * CPAD * 4);
  float*    Ssq    = (float*)   take((size_t)NN * TT * CPAD * 4);
  unsigned* Smin   = (unsigned*)take((size_t)NN * TT * CPAD * 4);
  unsigned* Smax   = (unsigned*)take((size_t)NN * TT * CPAD * 4);
  float*    outpre = (float*)   take((size_t)NN * FIN * 4);
  float*    bnsum  = (float*)   take(FIN * 4);
  float*    bnsumsq= (float*)   take(FIN * 4);
  float*    bnmu   = (float*)   take(FIN * 4);
  float*    bnrstd = (float*)   take(FIN * 4);
  float*    deg    = (float*)   take((size_t)NN * 4);
  float*    avgacc = (float*)   take(256);
  float*    pooled = (float*)   take((size_t)GG * FIN * 4);

  const long NAGG = (long)NN * TT * CPAD;

  // stage 0
  k_fill<<<gdiv(NN, 256), 256, 0, stream>>>(deg, NN, 0.f);
  k_fill<<<1, 256, 0, stream>>>(avgacc, 1, 0.f);
  k_fill<<<gdiv((long)GG * FIN, 256), 256, 0, stream>>>(pooled, (long)GG * FIN, 0.f);
  k_gather_h0<<<gdiv((long)NN * KPAD, 256), 256, 0, stream>>>(node_emb, x, h, hb);
  k_deg<<<gdiv(EE, 256), 256, 0, stream>>>(dstp, deg);
  k_avglog<<<gdiv(NN, 256), 256, 0, stream>>>(deg, avgacc);
  k_scale1<<<1, 1, 0, stream>>>(avgacc, 1.f / (float)NN);

  for (int l = 0; l < 4; ++l) {
    const float* We_l    = We    + (size_t)l * ED * FIN;
    const float* be_l    = be    + (size_t)l * FIN;
    const float* Wpre_l  = Wpre  + (size_t)l * TT * (3 * FIN) * FIN;
    const float* bpre_l  = bpre  + (size_t)l * TT * FIN;
    const float* Wpost_l = Wpost + (size_t)l * TT * (13 * FIN) * FOUT;
    const float* bpost_l = bpost + (size_t)l * TT * FOUT;
    const float* Wlin_l  = Wlin  + (size_t)l * FIN * FIN;
    const float* blin_l  = blin  + (size_t)l * FIN;
    const float* gam_l   = gam   + (size_t)l * FIN;
    const float* bet_l   = bet   + (size_t)l * FIN;

    k_fill<<<gdiv(NAGG, 256), 256, 0, stream>>>(Ssum, NAGG, 0.f);
    k_fill<<<gdiv(NAGG, 256), 256, 0, stream>>>(Ssq, NAGG, 0.f);
    k_fillu<<<gdiv(NAGG, 256), 256, 0, stream>>>(Smin, NAGG, 0xFFFFFFFFu); // key(+inf)
    k_fillu<<<gdiv(NAGG, 256), 256, 0, stream>>>(Smax, NAGG, 0x00000000u); // key(-inf)
    k_fill<<<1, 128, 0, stream>>>(bnsum, FIN, 0.f);
    k_fill<<<1, 128, 0, stream>>>(bnsumsq, FIN, 0.f);

    k_prep_w<<<gdiv(2L * TT * NT5 * 3 * 32 * 16, 256), 256, 0, stream>>>(Wpre_l, Wbs);
    k_c3<<<1, 256, 0, stream>>>(edge_emb, We_l, be_l, Wpre_l, bpre_l, c3);

    dim3 eg(125, NTILES);   // 125 blocks x 8 waves per (tower, N-tile); 25 N-tiles
    k_edge_msgs<<<eg, 256, 0, stream>>>(hb, Wbs, c3, srcp, dstp, eattr,
                                        Ssum, Ssq, Smin, Smax);

    k_node_post<<<NN, 128, 0, stream>>>(h, Ssum, Ssq, Smin, Smax, deg, avgacc, Wpost_l,
                                        bpost_l, Wlin_l, blin_l, outpre, bnsum, bnsumsq);
    k_bnstats<<<1, 128, 0, stream>>>(bnsum, bnsumsq, bnmu, bnrstd);
    k_bn_act<<<gdiv((long)NN * FIN, 256), 256, 0, stream>>>(outpre, bnmu, bnrstd, gam_l,
                                                            bet_l, uafp, h, hb);
  }

  k_pool<<<gdiv((long)NN * FIN, 256), 256, 0, stream>>>(h, batch, pooled);
  k_mlp<<<GG, 64, 0, stream>>>(pooled, mW1, mb1, mW2, mb2, mW3, mb3, uafp, out);
}
